// S4Block_66889820668457
// MI455X (gfx1250) — compile-verified, matrix operand order checked by reading, not run on hardware
//
#include <hip/hip_runtime.h>
#include <hip/hip_bf16.h>
#include <stdint.h>

// ---------------- problem dims ----------------
#define B_SZ     4
#define L_SEQ    2048
#define D_MODEL  1024
#define D_STATE  16
#define M_TOK    (B_SZ * L_SEQ)      // 8192 rows
#define K_DIM    1024                // K for both GEMMs
#define NKSTEP   (K_DIM / 32)        // 32 k-steps of BK=32

// ---------------- WMMA types ----------------
typedef __attribute__((ext_vector_type(16))) __bf16 v16bf;
typedef __attribute__((ext_vector_type(8)))  float  v8f;
typedef __attribute__((ext_vector_type(4)))  int    v4i_t;

union FragBF { v16bf v; uint4 q[2]; };
union AccF   { v8f  v; float f[8]; };

#define AS1 __attribute__((address_space(1)))
#define AS3 __attribute__((address_space(3)))

#if defined(__has_builtin)
#  if __has_builtin(__builtin_amdgcn_global_load_async_to_lds_b128)
#    define HAVE_ASYNC_LDS 1
#  endif
#endif
#ifndef HAVE_ASYNC_LDS
#  define HAVE_ASYNC_LDS 0
#endif

__device__ __forceinline__ void wait_async_all() {
#if HAVE_ASYNC_LDS
#  if __has_builtin(__builtin_amdgcn_s_wait_asynccnt)
  __builtin_amdgcn_s_wait_asynccnt(0);
#  else
  asm volatile("s_wait_asynccnt 0" ::: "memory");
#  endif
#endif
}

__device__ __forceinline__ unsigned short f2bf(float f) {
  unsigned int u = __float_as_uint(f);
  u = (u + 0x7FFFu + ((u >> 16) & 1u)) >> 16;  // round-to-nearest-even
  return (unsigned short)u;
}

// ---------------- tile staging: global -> LDS ----------------
// 128x32 bf16 tiles of A and B: 512 16B chunks each, 2 per thread per matrix.
#define LDSS 40        // padded LDS row stride in ushorts (80 B, 16B aligned)
#define TILE_USH (128 * LDSS)

__device__ __forceinline__ void stage_tile(
    const unsigned short* __restrict__ A, const unsigned short* __restrict__ B,
    int blockM, int blockN, int kk,
    unsigned short* sA, unsigned short* sB, int tid)
{
  const int c0 = tid, c1 = tid + 256;
  const int r0 = c0 >> 2, q0 = c0 & 3;
  const int r1 = c1 >> 2, q1 = c1 & 3;
  const unsigned short* ga0 = A + (size_t)(blockM * 128 + r0) * K_DIM + kk + q0 * 8;
  const unsigned short* ga1 = A + (size_t)(blockM * 128 + r1) * K_DIM + kk + q1 * 8;
  const unsigned short* gb0 = B + (size_t)(blockN * 128 + r0) * K_DIM + kk + q0 * 8;
  const unsigned short* gb1 = B + (size_t)(blockN * 128 + r1) * K_DIM + kk + q1 * 8;
  unsigned short* la0 = sA + r0 * LDSS + q0 * 8;
  unsigned short* la1 = sA + r1 * LDSS + q1 * 8;
  unsigned short* lb0 = sB + r0 * LDSS + q0 * 8;
  unsigned short* lb1 = sB + r1 * LDSS + q1 * 8;
#if HAVE_ASYNC_LDS
  // Direct global -> LDS async copies (ASYNCcnt), no VGPR round trip.
  __builtin_amdgcn_global_load_async_to_lds_b128((AS1 v4i_t*)ga0, (AS3 v4i_t*)la0, 0, 0);
  __builtin_amdgcn_global_load_async_to_lds_b128((AS1 v4i_t*)ga1, (AS3 v4i_t*)la1, 0, 0);
  __builtin_amdgcn_global_load_async_to_lds_b128((AS1 v4i_t*)gb0, (AS3 v4i_t*)lb0, 0, 0);
  __builtin_amdgcn_global_load_async_to_lds_b128((AS1 v4i_t*)gb1, (AS3 v4i_t*)lb1, 0, 0);
#else
  *(uint4*)la0 = *(const uint4*)ga0;
  *(uint4*)la1 = *(const uint4*)ga1;
  *(uint4*)lb0 = *(const uint4*)gb0;
  *(uint4*)lb1 = *(const uint4*)gb1;
#endif
}

// ---------------- shared WMMA mainloop (double-buffered) ----------------
// Block tile 128(M) x 128(N), 8 wave32 waves: waveM in [0,4), waveN in [0,2),
// each wave owns 32x64 = 2 Mtiles x 4 Ntiles of 16x16 (8 WMMA / k-step).
__device__ __forceinline__ void wmma_tile_gemm(
    const unsigned short* __restrict__ A,   // [*, 1024] bf16, K contiguous
    const unsigned short* __restrict__ B,   // [*, 1024] bf16, row n = output col n
    int blockM, int blockN,
    unsigned short* sA, unsigned short* sB, // each 2 * TILE_USH (double buffer)
    AccF (&acc)[2][4])
{
  const int tid   = threadIdx.x;
  const int lane  = tid & 31;
  const int wid   = tid >> 5;
  const int waveM = wid >> 1;
  const int waveN = wid & 1;
  const int half8 = (lane >> 4) << 3;   // 0 or 8 (per-lane K sub-block)
  const int ln16  = lane & 15;

  #pragma unroll
  for (int i = 0; i < 2; ++i)
    #pragma unroll
    for (int j = 0; j < 4; ++j)
      #pragma unroll
      for (int e = 0; e < 8; ++e) acc[i][j].f[e] = 0.0f;

  // prologue: stage tile 0 into buffer 0
  stage_tile(A, B, blockM, blockN, 0, sA, sB, tid);
  wait_async_all();
  __syncthreads();

  for (int k = 0; k < NKSTEP; ++k) {
    const int cur = k & 1;
    unsigned short* cA = sA + cur * TILE_USH;
    unsigned short* cB = sB + cur * TILE_USH;
    // issue next tile into the other buffer while computing this one
    if (k + 1 < NKSTEP)
      stage_tile(A, B, blockM, blockN, (k + 1) * 32,
                 sA + (cur ^ 1) * TILE_USH, sB + (cur ^ 1) * TILE_USH, tid);

    // Fragment gather (16-bit A/B layout): lane holds row/col m=l%16,
    // K elements i=0..15 at k = (i/8)*16 + (l/16)*8 + (i%8) -> two b128 reads.
    FragBF af[2], bfg[4];
    #pragma unroll
    for (int i = 0; i < 2; ++i) {
      const unsigned short* p = cA + (size_t)(waveM * 32 + i * 16 + ln16) * LDSS + half8;
      af[i].q[0] = *(const uint4*)(p);
      af[i].q[1] = *(const uint4*)(p + 16);
    }
    #pragma unroll
    for (int j = 0; j < 4; ++j) {
      const unsigned short* p = cB + (size_t)(waveN * 64 + j * 16 + ln16) * LDSS + half8;
      bfg[j].q[0] = *(const uint4*)(p);
      bfg[j].q[1] = *(const uint4*)(p + 16);
    }
    #pragma unroll
    for (int i = 0; i < 2; ++i)
      #pragma unroll
      for (int j = 0; j < 4; ++j)
        acc[i][j].v = __builtin_amdgcn_wmma_f32_16x16x32_bf16(
            false, af[i].v, false, bfg[j].v, (short)0, acc[i][j].v, false, false);

    wait_async_all();   // next-tile LDS writes complete (this wave)
    __syncthreads();    // visible to all waves; all reads of 'cur' done
  }
}

// ---------------- kernel 1: LayerNorm -> bf16 ----------------
__global__ __launch_bounds__(256) void s4_ln_kernel(
    const float* __restrict__ x, const float* __restrict__ g,
    const float* __restrict__ b, unsigned short* __restrict__ xn)
{
  const int row = blockIdx.x;
  const int tid = threadIdx.x;
  const float* xr = x + (size_t)row * D_MODEL;
  float v[4], s = 0.f, s2 = 0.f;
  #pragma unroll
  for (int i = 0; i < 4; ++i) {
    v[i] = xr[tid + i * 256];
    s += v[i]; s2 += v[i] * v[i];
  }
  __shared__ float rs[256], rs2[256];
  rs[tid] = s; rs2[tid] = s2;
  __syncthreads();
  for (int off = 128; off > 0; off >>= 1) {
    if (tid < off) { rs[tid] += rs[tid + off]; rs2[tid] += rs2[tid + off]; }
    __syncthreads();
  }
  const float mu  = rs[0] * (1.0f / D_MODEL);
  const float var = rs2[0] * (1.0f / D_MODEL) - mu * mu;
  const float inv = rsqrtf(var + 1e-5f);
  #pragma unroll
  for (int i = 0; i < 4; ++i) {
    int c = tid + i * 256;
    float xv = (v[i] - mu) * inv * g[c] + b[c];
    xn[(size_t)row * D_MODEL + c] = f2bf(xv);
  }
}

// ---------------- kernel 2: f32 -> bf16 conversion ----------------
__global__ __launch_bounds__(256) void s4_cvt_bf16_kernel(
    const float* __restrict__ src, unsigned short* __restrict__ dst, int n)
{
  int i = blockIdx.x * 256 + threadIdx.x;
  if (i < n) dst[i] = f2bf(src[i]);
}

// ---------------- kernel 3: GEMM1 (xz = xn @ in_w^T + in_b) ----------------
__global__ __launch_bounds__(256) void s4_gemm_xz_kernel(
    const unsigned short* __restrict__ Abf,   // xn bf16 [8192][1024]
    const unsigned short* __restrict__ Wbf,   // in_w bf16 [2048][1024]
    const float* __restrict__ in_b,
    float* __restrict__ xproj, float* __restrict__ zsilu)
{
  __shared__ __align__(16) unsigned short sA[2 * TILE_USH];
  __shared__ __align__(16) unsigned short sB[2 * TILE_USH];
  AccF acc[2][4];
  const int blockN = blockIdx.x, blockM = blockIdx.y;
  wmma_tile_gemm(Abf, Wbf, blockM, blockN, sA, sB, acc);

  const int lane  = threadIdx.x & 31;
  const int wid   = threadIdx.x >> 5;
  const int waveM = wid >> 1, waveN = wid & 1;
  const int half8 = (lane >> 4) << 3, ln16 = lane & 15;
  #pragma unroll
  for (int i = 0; i < 2; ++i)
    #pragma unroll
    for (int j = 0; j < 4; ++j) {
      const int gn = blockN * 128 + waveN * 64 + j * 16 + ln16;
      const float bias = in_b[gn];
      #pragma unroll
      for (int r = 0; r < 8; ++r) {
        const int gm = blockM * 128 + waveM * 32 + i * 16 + half8 + r;
        float val = acc[i][j].f[r] + bias;
        if (gn < D_MODEL) {
          xproj[(size_t)gm * D_MODEL + gn] = val;
        } else {
          zsilu[(size_t)gm * D_MODEL + (gn - D_MODEL)] = val / (1.0f + __expf(-val));
        }
      }
    }
}

// ---------------- kernel 4: SSM recurrence (replaces FFT conv) ----------------
// y[l] = sum_n C[n] * h[l,n];  h[l,n] = Abar[n]*h[l-1,n] + dt*B[n]*u[l]
__global__ __launch_bounds__(256) void s4_scan_kernel(
    const float* __restrict__ xproj, const float* __restrict__ zsilu,
    const float* __restrict__ A_log, const float* __restrict__ B_param,
    const float* __restrict__ C_param, const float* __restrict__ log_dt,
    unsigned short* __restrict__ ygate)
{
  const int t = blockIdx.x * 256 + threadIdx.x;   // 0..4095
  const int b = t >> 10;
  const int d = t & (D_MODEL - 1);
  float dt = __expf(log_dt[d]);
  dt = fminf(fmaxf(dt, 1e-4f), 1.0f);

  float Abar[D_STATE], dtB[D_STATE], Cc[D_STATE], h[D_STATE];
  #pragma unroll
  for (int n = 0; n < D_STATE; ++n) {
    float ab = __expf(-__expf(A_log[d * D_STATE + n]) * dt);
    ab = fminf(fmaxf(ab, 1e-8f), 1.0f - 1e-8f);
    Abar[n] = ab;
    dtB[n]  = B_param[d * D_STATE + n] * dt;
    Cc[n]   = C_param[d * D_STATE + n];
    h[n]    = 0.0f;
  }
  const size_t base = (size_t)b * L_SEQ * D_MODEL + d;
  for (int l = 0; l < L_SEQ; ++l) {
    const size_t idx = base + (size_t)l * D_MODEL;
    const float u = xproj[idx];
    float y = 0.0f;
    #pragma unroll
    for (int n = 0; n < D_STATE; ++n) {
      h[n] = fmaf(Abar[n], h[n], dtB[n] * u);
      y    = fmaf(Cc[n], h[n], y);
    }
    ygate[idx] = f2bf(y * zsilu[idx]);
  }
}

// ---------------- kernel 5: GEMM2 (out = clip(yg @ out_w^T + out_b + resid)) ----
__global__ __launch_bounds__(256) void s4_gemm_out_kernel(
    const unsigned short* __restrict__ Ybf,   // ygate bf16 [8192][1024]
    const unsigned short* __restrict__ Wbf,   // out_w bf16 [1024][1024]
    const float* __restrict__ out_b, const float* __restrict__ resid,
    float* __restrict__ out)
{
  __shared__ __align__(16) unsigned short sA[2 * TILE_USH];
  __shared__ __align__(16) unsigned short sB[2 * TILE_USH];
  AccF acc[2][4];
  const int blockN = blockIdx.x, blockM = blockIdx.y;
  wmma_tile_gemm(Ybf, Wbf, blockM, blockN, sA, sB, acc);

  const int lane  = threadIdx.x & 31;
  const int wid   = threadIdx.x >> 5;
  const int waveM = wid >> 1, waveN = wid & 1;
  const int half8 = (lane >> 4) << 3, ln16 = lane & 15;
  #pragma unroll
  for (int i = 0; i < 2; ++i)
    #pragma unroll
    for (int j = 0; j < 4; ++j) {
      const int gn = blockN * 128 + waveN * 64 + j * 16 + ln16;
      const float bias = out_b[gn];
      #pragma unroll
      for (int r = 0; r < 8; ++r) {
        const int gm = blockM * 128 + waveM * 32 + i * 16 + half8 + r;
        float val = acc[i][j].f[r] + bias + resid[(size_t)gm * D_MODEL + gn];
        val = fminf(10.0f, fmaxf(-10.0f, val));
        out[(size_t)gm * D_MODEL + gn] = val;
      }
    }
}

// ---------------- launcher ----------------
extern "C" void kernel_launch(void* const* d_in, const int* in_sizes, int n_in,
                              void* d_out, int out_size, void* d_ws, size_t ws_size,
                              hipStream_t stream) {
  const float* x       = (const float*)d_in[0];
  const float* A_log   = (const float*)d_in[1];
  const float* B_param = (const float*)d_in[2];
  const float* C_param = (const float*)d_in[3];
  const float* log_dt  = (const float*)d_in[4];
  const float* in_w    = (const float*)d_in[5];
  const float* in_b    = (const float*)d_in[6];
  const float* out_w   = (const float*)d_in[7];
  const float* out_b   = (const float*)d_in[8];
  const float* ln_g    = (const float*)d_in[9];
  const float* ln_b    = (const float*)d_in[10];
  float* out = (float*)d_out;

  char* ws = (char*)d_ws;
  const size_t MB = 1024ull * 1024ull;
  unsigned short* xn_bf   = (unsigned short*)(ws + 0 * MB);    // 16 MB
  unsigned short* inw_bf  = (unsigned short*)(ws + 16 * MB);   //  4 MB
  unsigned short* outw_bf = (unsigned short*)(ws + 20 * MB);   //  2 MB
  float*          xproj   = (float*)(ws + 22 * MB);            // 32 MB
  float*          zsilu   = (float*)(ws + 54 * MB);            // 32 MB
  unsigned short* ygate   = (unsigned short*)(ws + 86 * MB);   // 16 MB

  // weights -> bf16
  s4_cvt_bf16_kernel<<<(2 * D_MODEL * D_MODEL) / 256, 256, 0, stream>>>(in_w, inw_bf, 2 * D_MODEL * D_MODEL);
  s4_cvt_bf16_kernel<<<(D_MODEL * D_MODEL) / 256, 256, 0, stream>>>(out_w, outw_bf, D_MODEL * D_MODEL);
  // LayerNorm -> bf16 activations
  s4_ln_kernel<<<M_TOK, 256, 0, stream>>>(x, ln_g, ln_b, xn_bf);
  // GEMM1: xz = xn @ in_w^T + in_b  (M=8192, N=2048, K=1024)
  s4_gemm_xz_kernel<<<dim3(2 * D_MODEL / 128, M_TOK / 128), 256, 0, stream>>>(
      xn_bf, inw_bf, in_b, xproj, zsilu);
  // SSM recurrence + gating
  s4_scan_kernel<<<(B_SZ * D_MODEL) / 256, 256, 0, stream>>>(
      xproj, zsilu, A_log, B_param, C_param, log_dt, ygate);
  // GEMM2: out = clip(yg @ out_w^T + out_b + resid, -10, 10)  (M=8192, N=1024, K=1024)
  s4_gemm_out_kernel<<<dim3(D_MODEL / 128, M_TOK / 128), 256, 0, stream>>>(
      ygate, outw_bf, out_b, x, out);
}
